// GraphConv_49108656063244
// MI455X (gfx1250) — matrix operand order repaired and checked by reading, not compiled, not run on hardware
//
#include <hip/hip_runtime.h>
#include <hip/hip_bf16.h>
#include <stdint.h>

typedef unsigned short ushort_t;
typedef __attribute__((ext_vector_type(16))) __bf16 v16bf;
typedef __attribute__((ext_vector_type(8)))  __bf16 v8bf;
typedef __attribute__((ext_vector_type(8)))  float  v8f;

#define F_DIM 256
#define NEG_SLOPE 0.01f
#define LN_EPS 1e-5f
#define KEEP_SCALE 1.25f               // 1 / 0.8
#define KEEP_THRESH 3435973837u        // 0.8 * 2^32

// ---------------- utility: f32 -> bf16 (round-to-nearest-even) ----------------
__device__ __forceinline__ ushort_t f2bf(float f) {
    uint32_t x = __float_as_uint(f);
    uint32_t r = x + 0x7fffu + ((x >> 16) & 1u);
    return (ushort_t)(r >> 16);
}

__device__ __forceinline__ uint32_t hashrng(uint32_t x) {
    x ^= x >> 17; x *= 0xed5ad4bbu;
    x ^= x >> 11; x *= 0xac4c1b51u;
    x ^= x >> 15; x *= 0x31848babu;
    x ^= x >> 14;
    return x;
}

// ---------------- kernel 0: zero agg (graph-capture-safe init) ----------------
__global__ void zero_f32_kernel(float* __restrict__ p, int n4) {
    int i = blockIdx.x * blockDim.x + threadIdx.x;
    if (i < n4) {
        float4 z = {0.f, 0.f, 0.f, 0.f};
        ((float4*)p)[i] = z;
    }
}

// ---------------- kernel 1: f32 -> bf16 conversion (2 elems / thread) ---------
__global__ void cvt_bf16_kernel(const float* __restrict__ in,
                                ushort_t* __restrict__ out, int n2) {
    int i = blockIdx.x * blockDim.x + threadIdx.x;
    if (i < n2) {
        float2 v = ((const float2*)in)[i];
        uint32_t packed = (uint32_t)f2bf(v.x) | ((uint32_t)f2bf(v.y) << 16);
        ((uint32_t*)out)[i] = packed;
    }
}

// ---------------- kernel 2: SpMM edge scatter (one wave32 per edge) ----------
// agg[dst[e]][:] += ev[e] * X[src[e]][:]   ; X and agg are L2-resident.
__global__ __launch_bounds__(256) void spmm_kernel(
        const int* __restrict__ src, const int* __restrict__ dst,
        const float* __restrict__ ev, const float* __restrict__ X,
        float* __restrict__ agg, int nE) {
    int e    = (blockIdx.x * blockDim.x + threadIdx.x) >> 5;
    int lane = threadIdx.x & 31;
    if (e >= nE) return;
    int   s = src[e];
    int   d = dst[e];
    float w = ev[e];
    const float4* xs = (const float4*)(X + (size_t)s * F_DIM);
    float* ad = agg + (size_t)d * F_DIM;
    #pragma unroll
    for (int i = 0; i < 2; ++i) {
        int    c4 = lane + 32 * i;       // float4 index 0..63
        float4 v  = xs[c4];
        int    c  = c4 * 4;
        __hip_atomic_fetch_add(ad + c + 0, w * v.x, __ATOMIC_RELAXED, __HIP_MEMORY_SCOPE_AGENT);
        __hip_atomic_fetch_add(ad + c + 1, w * v.y, __ATOMIC_RELAXED, __HIP_MEMORY_SCOPE_AGENT);
        __hip_atomic_fetch_add(ad + c + 2, w * v.z, __ATOMIC_RELAXED, __HIP_MEMORY_SCOPE_AGENT);
        __hip_atomic_fetch_add(ad + c + 3, w * v.w, __ATOMIC_RELAXED, __HIP_MEMORY_SCOPE_AGENT);
    }
}

// ---------------- kernel 3: WMMA GEMM + fused LayerNorm/LeakyReLU/Dropout ----
// Block = 16 output rows, all 256 cols. 8 waves; each wave does 2 N-tiles of
// 16x16 via v_wmma_f32_16x16x32_bf16 (K-loop of 8). Epilogue via LDS.
__global__ __launch_bounds__(256) void gemm_ln_kernel(
        const ushort_t* __restrict__ aggb,   // [N,256] bf16, row-major
        const ushort_t* __restrict__ Wb,     // [256,256] bf16, row-major (O,K)
        float* __restrict__ out) {
    __shared__ __align__(16) ushort_t sA[16][F_DIM];  // 8 KB: A tile (bf16)
    __shared__ __align__(16) float    sO[16][F_DIM];  // 16 KB: D tile (f32)

    const int tid  = threadIdx.x;
    const int row0 = blockIdx.x * 16;

    // --- stage A tile: 16 rows x 256 bf16 (each thread copies 32 bytes) ---
    {
        int r  = tid >> 4;              // 0..15
        int c0 = (tid & 15) * 16;       // 0..240
        const uint4* g = (const uint4*)(aggb + (size_t)(row0 + r) * F_DIM + c0);
        uint4* s = (uint4*)&sA[r][c0];
        s[0] = g[0];
        s[1] = g[1];
    }
    __syncthreads();

    const int wave = tid >> 5;
    const int lane = tid & 31;
    const int m    = lane & 15;         // A row within tile / B column within tile
    const int hi   = lane >> 4;         // lane half selects K sub-range

    #pragma unroll
    for (int t = 0; t < 2; ++t) {
        const int o0 = (wave + t * 8) * 16;           // N-tile base column
        const ushort_t* wrow = Wb + (size_t)(o0 + m) * F_DIM;
        v8f c = {};
        #pragma unroll
        for (int kb = 0; kb < 8; ++kb) {
            const int kbase = kb * 32;
            // A fragment (16x32 bf16): lane half hi -> K {kbase+16hi .. +15}
            union { v16bf v; v8bf h[2]; } A;
            A.h[0] = *(const v8bf*)&sA[m][kbase + hi * 16];
            A.h[1] = *(const v8bf*)&sA[m][kbase + hi * 16 + 8];
            // B fragment (32x16 bf16): lane half hi -> K {kbase+16hi .. +15}, col = o0+m
            union { v16bf v; v8bf h[2]; } B;
            const ushort_t* bp = wrow + kbase + hi * 16;
            B.h[0] = *(const v8bf*)(bp);
            B.h[1] = *(const v8bf*)(bp + 8);
            c = __builtin_amdgcn_wmma_f32_16x16x32_bf16(
                    false, A.v, false, B.v, (short)0, c, false, false);
        }
        // D layout: VGPR r, lane -> row (r + 8*hi), col (o0 + m)
        #pragma unroll
        for (int r = 0; r < 8; ++r) {
            sO[r + hi * 8][o0 + m] = c[r];
        }
    }
    __syncthreads();

    // --- fused LayerNorm + LeakyReLU + dropout: wave handles 2 rows ---
    #pragma unroll
    for (int rr = 0; rr < 2; ++rr) {
        const int row = wave * 2 + rr;
        float vals[8];
        float s = 0.f, s2 = 0.f;
        #pragma unroll
        for (int i = 0; i < 8; ++i) {          // conflict-free: col = lane + 32*i
            float v = sO[row][lane + 32 * i];
            vals[i] = v;
            s  += v;
            s2 += v * v;
        }
        #pragma unroll
        for (int off = 16; off; off >>= 1) {
            s  += __shfl_xor(s,  off, 32);
            s2 += __shfl_xor(s2, off, 32);
        }
        const float mu  = s * (1.0f / F_DIM);
        const float var = fmaxf(s2 * (1.0f / F_DIM) - mu * mu, 0.f);
        const float rs  = rsqrtf(var + LN_EPS);
        const size_t grow = (size_t)(row0 + row) * F_DIM;
        #pragma unroll
        for (int i = 0; i < 8; ++i) {
            const int col = lane + 32 * i;
            float v = (vals[i] - mu) * rs;
            v = (v >= 0.f) ? v : NEG_SLOPE * v;
            uint32_t h = hashrng((uint32_t)(grow + col) * 2654435761u + 42u);
            v = (h < KEEP_THRESH) ? v * KEEP_SCALE : 0.f;
            out[grow + col] = v;
        }
    }
}

// ---------------- launcher ----------------
extern "C" void kernel_launch(void* const* d_in, const int* in_sizes, int n_in,
                              void* d_out, int out_size, void* d_ws, size_t ws_size,
                              hipStream_t stream) {
    const int*   src = (const int*)  d_in[0];
    const int*   dst = (const int*)  d_in[1];
    const float* ev  = (const float*)d_in[2];
    const float* X   = (const float*)d_in[3];
    const float* W   = (const float*)d_in[4];
    float* out = (float*)d_out;

    const int nE = in_sizes[0];
    const int nN = in_sizes[3] / F_DIM;          // 100000
    const int nW = in_sizes[4];                  // 65536

    // workspace layout (all 256B-aligned by construction)
    char* ws = (char*)d_ws;
    float*    agg_f32  = (float*)ws;                                     // N*256*4
    size_t    agg_f32_bytes = (size_t)nN * F_DIM * sizeof(float);
    ushort_t* agg_bf16 = (ushort_t*)(ws + agg_f32_bytes);                // N*256*2
    size_t    agg_bf16_bytes = (size_t)nN * F_DIM * sizeof(ushort_t);
    ushort_t* W_bf16   = (ushort_t*)(ws + agg_f32_bytes + agg_bf16_bytes); // 128KB
    (void)ws_size; (void)n_in; (void)out_size;

    // 0) zero the f32 accumulator
    {
        int n4 = (nN * F_DIM) / 4;
        zero_f32_kernel<<<(n4 + 255) / 256, 256, 0, stream>>>(agg_f32, n4);
    }
    // 1) convert W to bf16 (L2-resident thereafter)
    {
        int n2 = nW / 2;
        cvt_bf16_kernel<<<(n2 + 255) / 256, 256, 0, stream>>>(W, W_bf16, n2);
    }
    // 2) SpMM edge scatter: one wave per edge, 8 waves per block
    {
        int blocks = (nE + 7) / 8;
        spmm_kernel<<<blocks, 256, 0, stream>>>(src, dst, ev, X, agg_f32, nE);
    }
    // 3) convert agg to bf16
    {
        int n2 = (nN * F_DIM) / 2;
        cvt_bf16_kernel<<<(n2 + 255) / 256, 256, 0, stream>>>(agg_f32, agg_bf16, n2);
    }
    // 4) WMMA GEMM + fused LayerNorm/LeakyReLU/Dropout (N = 6250*16 exactly)
    {
        int blocks = nN / 16;
        gemm_ln_kernel<<<blocks, 256, 0, stream>>>(agg_bf16, W_bf16, out);
    }
}
